// Attention_18562848653411
// MI455X (gfx1250) — matrix operand order repaired and checked by reading, not compiled, not run on hardware
//
#include <hip/hip_runtime.h>
#include <hip/hip_bf16.h>
#include <math.h>

// ---------------------------------------------------------------------------
// MI455X / gfx1250 implementation of the 4-layer attention-ish block.
// bf16 WMMA (v_wmma_f32_16x16x32_bf16) for all GEMMs, fp32 accumulate, fused
// epilogues (bias / GELU / 1/C scale / residual). 128x128x32 LDS tiles.
// Data movement: double-buffered GLOBAL_LOAD_ASYNC_TO_LDS (ASYNCcnt) when the
// toolchain exposes the gfx1250 async builtins; sync global->VGPR->LDS
// fallback otherwise. Workload is matrix-pipe bound (~584 GFLOP vs ~6us of
// HBM traffic at 23.3 TB/s), so everything funnels through WMMA.
// ---------------------------------------------------------------------------

typedef __attribute__((ext_vector_type(16))) __bf16        v16bf;
typedef __attribute__((ext_vector_type(8)))  float         v8f;
typedef __attribute__((ext_vector_type(8)))  unsigned int  v8u;
typedef int v4i_vec __attribute__((vector_size(4 * sizeof(int))));

#if defined(__has_builtin)
#if __has_builtin(__builtin_amdgcn_global_load_async_to_lds_b128) && \
    __has_builtin(__builtin_amdgcn_s_wait_asynccnt)
#define USE_ASYNC_LDS 1
#endif
#endif
#ifndef USE_ASYNC_LDS
#define USE_ASYNC_LDS 0
#endif

__device__ __forceinline__ unsigned short f2bf(float f) {
  unsigned int u = __builtin_bit_cast(unsigned int, f);
  u += 0x7FFFu + ((u >> 16) & 1u);            // round-to-nearest-even
  return (unsigned short)(u >> 16);
}
__device__ __forceinline__ float bf2f(unsigned short h) {
  unsigned int u = ((unsigned int)h) << 16;
  return __builtin_bit_cast(float, u);
}

#if USE_ASYNC_LDS
// 16B per lane, global -> LDS, tracked by ASYNCcnt (no VGPR round-trip).
// Builtin params are (v4i addrspace(1)*, v4i addrspace(3)*, imm offset, imm cpol).
__device__ __forceinline__ void async_cp16(const void* g, void* l) {
  __builtin_amdgcn_global_load_async_to_lds_b128(
      (__attribute__((address_space(1))) v4i_vec*)(g),
      (__attribute__((address_space(3))) v4i_vec*)(l), 0, 0);
}
#endif

// ------------------------------ fp32 -> bf16 --------------------------------
__global__ void k_f32_to_bf16(const float* __restrict__ in,
                              unsigned short* __restrict__ out, long long n) {
  long long i = (long long)blockIdx.x * blockDim.x + threadIdx.x;
  if (i < n) out[i] = f2bf(in[i]);
}

// ------------------------------ WMMA GEMM -----------------------------------
// C[M,N] = epilogue( A[M,K] @ op(B) + bias ), A bf16 row-major (lda),
//   B_NT=true : B is (N,K) row-major (weight / K^T style), ldb = row stride
//   B_NT=false: B is (K,N) row-major,                      ldb = row stride
// Tile 128x128x32, 256 threads (8 wave32: 2 M-waves x 4 N-waves, 64x32 each).
// All M,N,K used here are multiples of 128/32 -> no bounds checks.
#define BM 128
#define BN 128
#define BK 32
#define LSTR 40   // LDS row stride in halves: 80B rows -> 16B-aligned b128 ops

template <bool B_NT>
__global__ __launch_bounds__(256) void k_gemm_bf16(
    const unsigned short* __restrict__ A, long long sAb, int lda,
    const unsigned short* __restrict__ Bm, long long sBb, int ldb,
    void* __restrict__ Cv, long long sCb, int ldc,
    const float* __restrict__ bias, int M, int N, int K, float scale,
    int doGelu, int outBf16,
    unsigned short* __restrict__ predW,        // layer0: also store result here
    const unsigned short* __restrict__ predR)  // layers>0: add this residual
{
  const int tid   = threadIdx.x;
  const int lane  = tid & 31;
  const int wave  = tid >> 5;
  const int waveM = wave >> 2;   // 0..1
  const int waveN = wave & 3;    // 0..3
  const int lo    = lane & 15;
  const int hi    = lane >> 4;

  const int mBase = blockIdx.y * BM;
  const int nBase = blockIdx.x * BN;
  const long long zb = blockIdx.z;
  A  += zb * sAb;
  Bm += zb * sBb;

  // tile-fill assignments
  const int ldRow = tid >> 1;          // 0..127
  const int ldCol = (tid & 1) << 4;    // 0 or 16 (halves)
  const int bK    = tid >> 3;          // 0..31   (NN path)
  const int bN    = (tid & 7) << 4;    // 0..112  (NN path)

  v8f acc[4][2];
  #pragma unroll
  for (int i = 0; i < 4; ++i)
    #pragma unroll
    for (int j = 0; j < 2; ++j)
      #pragma unroll
      for (int r = 0; r < 8; ++r) acc[i][j][r] = 0.0f;

#if USE_ASYNC_LDS
  // ---- double-buffered async pipeline: copy tile k+1 while computing k ----
  __shared__ unsigned short lA[2][BM * LSTR];
  __shared__ unsigned short lB[2][BN * LSTR];

  auto stageA = [&](int k0, int buf) {
    const unsigned short* src = A + (long long)(mBase + ldRow) * lda + k0 + ldCol;
    unsigned short* dst = &lA[buf][ldRow * LSTR + ldCol];
    async_cp16(src, dst);
    async_cp16(src + 8, dst + 8);
  };
  auto stageB = [&](int k0, int buf) {
    if (B_NT) {
      const unsigned short* src = Bm + (long long)(nBase + ldRow) * ldb + k0 + ldCol;
      unsigned short* dst = &lB[buf][ldRow * LSTR + ldCol];
      async_cp16(src, dst);
      async_cp16(src + 8, dst + 8);
    } else {
      // (K,N) row-major: coalesced read along n, transpose-scatter into LDS
      const unsigned short* src = Bm + (long long)(k0 + bK) * ldb + nBase + bN;
      unsigned short tmp[16];
      *(uint4*)(tmp)     = *(const uint4*)(src);
      *(uint4*)(tmp + 8) = *(const uint4*)(src + 8);
      #pragma unroll
      for (int j = 0; j < 16; ++j) lB[buf][(bN + j) * LSTR + bK] = tmp[j];
    }
  };

  stageA(0, 0);
  stageB(0, 0);
#else
  __shared__ unsigned short lA[1][BM * LSTR];
  __shared__ unsigned short lB[1][BN * LSTR];
#endif

  for (int k0 = 0; k0 < K; k0 += BK) {
#if USE_ASYNC_LDS
    const int buf = (k0 / BK) & 1;
    if (k0 + BK < K) {
      stageA(k0 + BK, buf ^ 1);
      stageB(k0 + BK, buf ^ 1);
      // keep only the next tile's own-wave async copies outstanding
      __builtin_amdgcn_s_wait_asynccnt(B_NT ? 4 : 2);
    } else {
      __builtin_amdgcn_s_wait_asynccnt(0);
    }
    __syncthreads();
#else
    const int buf = 0;
    { // A tile (128x32)
      const unsigned short* src = A + (long long)(mBase + ldRow) * lda + k0 + ldCol;
      uint4 p0 = *(const uint4*)(src);
      uint4 p1 = *(const uint4*)(src + 8);
      *(uint4*)(&lA[0][ldRow * LSTR + ldCol])     = p0;
      *(uint4*)(&lA[0][ldRow * LSTR + ldCol + 8]) = p1;
      __builtin_prefetch(src + BK, 0, 1);   // speculative; OOB drops silently
    }
    if (B_NT) {
      const unsigned short* src = Bm + (long long)(nBase + ldRow) * ldb + k0 + ldCol;
      uint4 p0 = *(const uint4*)(src);
      uint4 p1 = *(const uint4*)(src + 8);
      *(uint4*)(&lB[0][ldRow * LSTR + ldCol])     = p0;
      *(uint4*)(&lB[0][ldRow * LSTR + ldCol + 8]) = p1;
      __builtin_prefetch(src + BK, 0, 1);
    } else {
      const unsigned short* src = Bm + (long long)(k0 + bK) * ldb + nBase + bN;
      unsigned short tmp[16];
      *(uint4*)(tmp)     = *(const uint4*)(src);
      *(uint4*)(tmp + 8) = *(const uint4*)(src + 8);
      #pragma unroll
      for (int j = 0; j < 16; ++j) lB[0][(bN + j) * LSTR + bK] = tmp[j];
      __builtin_prefetch(src + (long long)BK * ldb, 0, 1);
    }
    __syncthreads();
#endif

    // ---- build fragments (documented 16-bit A 16x32 / B 32x16 lane layouts) ----
    v16bf afr[4];
    #pragma unroll
    for (int i = 0; i < 4; ++i) {
      const int mf = waveM * 64 + i * 16;
      v8u ar;
      #pragma unroll
      for (int j = 0; j < 4; ++j) {            // VGPR0..3: K = hi*8 + 2j
        const int kk = hi * 8 + 2 * j;
        ar[j] = *(const unsigned int*)&lA[buf][(mf + lo) * LSTR + kk];
      }
      #pragma unroll
      for (int j = 0; j < 4; ++j) {            // VGPR4..7: K = 16 + hi*8 + 2j
        const int kk = 16 + hi * 8 + 2 * j;
        ar[4 + j] = *(const unsigned int*)&lA[buf][(mf + lo) * LSTR + kk];
      }
      afr[i] = __builtin_bit_cast(v16bf, ar);
    }
    v16bf bfr[2];
    #pragma unroll
    for (int j = 0; j < 2; ++j) {
      const int nf = waveN * 32 + j * 16;
      v8u br;
      #pragma unroll
      for (int v = 0; v < 8; ++v) {            // VGPR0..3: K = hi*8 + 2v, VGPR4..7: K = 16 + hi*8 + 2(v-4)
        const int kk = (v >> 2) * 16 + hi * 8 + 2 * (v & 3);
        br[v] = *(const unsigned int*)&lB[buf][(nf + lo) * LSTR + kk];
      }
      bfr[j] = __builtin_bit_cast(v16bf, br);
    }

    // ---- 8 WMMAs per wave per K-step ----
    #pragma unroll
    for (int i = 0; i < 4; ++i)
      #pragma unroll
      for (int j = 0; j < 2; ++j)
        acc[i][j] = __builtin_amdgcn_wmma_f32_16x16x32_bf16(
            false, afr[i], false, bfr[j], (short)0, acc[i][j], false, false);

    __syncthreads();
  }

  // ---- fused epilogue: scale -> +bias -> GELU -> residual -> store ----
  float*          Cf = (float*)Cv;
  unsigned short* Ch = (unsigned short*)Cv;
  const long long cOff = zb * sCb;
  #pragma unroll
  for (int i = 0; i < 4; ++i) {
    #pragma unroll
    for (int j = 0; j < 2; ++j) {
      const int col  = nBase + waveN * 32 + j * 16 + lo;
      const float bc = bias ? bias[col] : 0.0f;
      #pragma unroll
      for (int r = 0; r < 8; ++r) {            // C/D: VGPR r -> M = r + hi*8
        const int row = mBase + waveM * 64 + i * 16 + hi * 8 + r;
        float v = acc[i][j][r] * scale + bc;
        if (doGelu) v = 0.5f * v * (1.0f + erff(v * 0.70710678118654752f));
        const long long idx = (long long)row * ldc + col;
        if (predR) v += bf2f(predR[idx]);
        if (predW) predW[idx] = f2bf(v);
        if (outBf16) Ch[cOff + idx] = f2bf(v);
        else         Cf[cOff + idx] = v;
      }
    }
  }
}

// ------------------------------ LayerNorm -----------------------------------
__global__ __launch_bounds__(256) void k_layernorm_bf16(
    const float* __restrict__ x, const float* __restrict__ g,
    const float* __restrict__ b, unsigned short* __restrict__ out, int C) {
  __shared__ float red[256];
  const int row = blockIdx.x;
  const float* xr = x + (long long)row * C;
  float s = 0.f;
  for (int c = threadIdx.x; c < C; c += 256) s += xr[c];
  red[threadIdx.x] = s; __syncthreads();
  for (int o = 128; o > 0; o >>= 1) { if (threadIdx.x < o) red[threadIdx.x] += red[threadIdx.x + o]; __syncthreads(); }
  const float mu = red[0] / (float)C; __syncthreads();
  float v = 0.f;
  for (int c = threadIdx.x; c < C; c += 256) { float d = xr[c] - mu; v += d * d; }
  red[threadIdx.x] = v; __syncthreads();
  for (int o = 128; o > 0; o >>= 1) { if (threadIdx.x < o) red[threadIdx.x] += red[threadIdx.x + o]; __syncthreads(); }
  const float rstd = rsqrtf(red[0] / (float)C + 1e-5f);
  unsigned short* orow = out + (long long)row * C;
  for (int c = threadIdx.x; c < C; c += 256)
    orow[c] = f2bf((xr[c] - mu) * rstd * g[c] + b[c]);
}

// ------------------------------ Softmax (rows) ------------------------------
__global__ __launch_bounds__(256) void k_softmax_bf16(
    const float* __restrict__ x, unsigned short* __restrict__ out, int C) {
  __shared__ float red[256];
  const int row = blockIdx.x;
  const float* xr = x + (long long)row * C;
  float mx = -3.0e38f;
  for (int c = threadIdx.x; c < C; c += 256) mx = fmaxf(mx, xr[c]);
  red[threadIdx.x] = mx; __syncthreads();
  for (int o = 128; o > 0; o >>= 1) { if (threadIdx.x < o) red[threadIdx.x] = fmaxf(red[threadIdx.x], red[threadIdx.x + o]); __syncthreads(); }
  const float m = red[0]; __syncthreads();
  float s = 0.f;
  for (int c = threadIdx.x; c < C; c += 256) s += expf(xr[c] - m);
  red[threadIdx.x] = s; __syncthreads();
  for (int o = 128; o > 0; o >>= 1) { if (threadIdx.x < o) red[threadIdx.x] += red[threadIdx.x + o]; __syncthreads(); }
  const float inv = 1.0f / red[0];
  unsigned short* orow = out + (long long)row * C;
  for (int c = threadIdx.x; c < C; c += 256)
    orow[c] = f2bf(expf(xr[c] - m) * inv);
}

// ------------------------------ pos-enc add ---------------------------------
__global__ void k_pose_add(float* __restrict__ out, int S, int C, long long n) {
  long long i = (long long)blockIdx.x * blockDim.x + threadIdx.x;
  if (i >= n) return;
  const int c = (int)(i % C);
  const int s = (int)((i / C) % S);
  const float div = powf(1000.0f, -2.0f * (float)c / (float)C);
  const float a = (float)s * div;
  out[i] += (s & 1) ? cosf(a) : sinf(a);
}

// ---------------------------------------------------------------------------
extern "C" void kernel_launch(void* const* d_in, const int* in_sizes, int n_in,
                              void* d_out, int out_size, void* d_ws, size_t ws_size,
                              hipStream_t stream) {
  (void)in_sizes; (void)n_in; (void)out_size; (void)ws_size;
  const int Bb = 4, S = 2048, C = 1024, Hh = 4;
  const int M = Bb * S;  // 8192

  const float* x        = (const float*)d_in[0];
  const float* fc_in_w  = (const float*)d_in[1];
  const float* fc_in_b  = (const float*)d_in[2];
  const float* ln_g     = (const float*)d_in[3];
  const float* ln_b     = (const float*)d_in[4];
  const float* qkv_w    = (const float*)d_in[5];
  const float* qkv_b    = (const float*)d_in[6];
  const float* proj_w   = (const float*)d_in[7];
  const float* proj_b   = (const float*)d_in[8];
  const float* fc_out_w = (const float*)d_in[9];
  const float* fc_out_b = (const float*)d_in[10];
  float* out = (float*)d_out;

  // workspace carve-up (~196 MB total, 256B aligned chunks)
  char* ws = (char*)d_ws;
  auto carve = [&](size_t bytes) -> char* {
    char* p = ws; ws += (bytes + 255) & ~(size_t)255; return p;
  };
  unsigned short* wFcIn  = (unsigned short*)carve((size_t)C * C * 2);
  unsigned short* wQkv   = (unsigned short*)carve((size_t)Hh * 3 * C * C * 2);
  unsigned short* wProj  = (unsigned short*)carve((size_t)Hh * C * C * 2);
  unsigned short* wFcOut = (unsigned short*)carve((size_t)C * C * 2);
  unsigned short* actA   = (unsigned short*)carve((size_t)M * C * 2);      // x_bf16, later softmax(att)
  unsigned short* hb     = (unsigned short*)carve((size_t)M * C * 2);      // running h (bf16)
  unsigned short* qkvB   = (unsigned short*)carve((size_t)M * 3 * C * 2);  // gelu(qkv)
  unsigned short* scores = (unsigned short*)carve((size_t)Bb * S * S * 2);
  float*          f32b   = (float*)carve((size_t)M * C * 4);               // pre-LN h, then raw att
  unsigned short* predB  = (unsigned short*)carve((size_t)M * C * 2);      // layer-0 residual

  auto cvt = [&](const float* src, unsigned short* dst, long long n) {
    k_f32_to_bf16<<<dim3((unsigned)((n + 255) / 256)), 256, 0, stream>>>(src, dst, n);
  };
  cvt(x,        actA,  (long long)M * C);
  cvt(fc_in_w,  wFcIn, (long long)C * C);
  cvt(qkv_w,    wQkv,  (long long)Hh * 3 * C * C);
  cvt(proj_w,   wProj, (long long)Hh * C * C);
  cvt(fc_out_w, wFcOut,(long long)C * C);

  // h = x @ fc_in_w^T + b   (fp32 out, then LayerNorm -> bf16)
  k_gemm_bf16<true><<<dim3(C / 128, M / 128, 1), 256, 0, stream>>>(
      actA, 0, C, wFcIn, 0, C, f32b, 0, C, fc_in_b, M, C, C,
      1.0f, /*gelu*/0, /*bf16out*/0, nullptr, nullptr);
  k_layernorm_bf16<<<M, 256, 0, stream>>>(f32b, ln_g, ln_b, hb, C);

  for (int i = 0; i < Hh; ++i) {
    // qkv = gelu(h @ qkv_w[i]^T + qkv_b[i])  -> bf16, row stride 3C
    k_gemm_bf16<true><<<dim3(3 * C / 128, M / 128, 1), 256, 0, stream>>>(
        hb, 0, C, wQkv + (size_t)i * 3 * C * C, 0, C, qkvB, 0, 3 * C,
        qkv_b + (size_t)i * 3 * C, M, 3 * C, C, 1.0f, 1, 1, nullptr, nullptr);

    // scores = (q @ k^T) / C   (batched over B; both operands K-contiguous = NT)
    k_gemm_bf16<true><<<dim3(S / 128, S / 128, Bb), 256, 0, stream>>>(
        qkvB, (long long)S * 3 * C, 3 * C,           // q  (col 0)
        qkvB + C, (long long)S * 3 * C, 3 * C,       // k  (col C)
        scores, (long long)S * S, S, nullptr,
        S, S, C, 1.0f / (float)C, 0, 1, nullptr, nullptr);

    // att = scores @ v   (NN: v is (t, c) row-major)  -> fp32 for softmax
    k_gemm_bf16<false><<<dim3(C / 128, S / 128, Bb), 256, 0, stream>>>(
        scores, (long long)S * S, S,
        qkvB + 2 * C, (long long)S * 3 * C, 3 * C,   // v  (col 2C)
        f32b, (long long)S * C, C, nullptr,
        S, C, S, 1.0f, 0, 0, nullptr, nullptr);

    // softmax over last axis -> bf16
    k_softmax_bf16<<<M, 256, 0, stream>>>(f32b, actA, C);

    // h = gelu(att @ proj_w[i]^T + proj_b[i]) (+pred for i>0; save pred at i==0)
    k_gemm_bf16<true><<<dim3(C / 128, M / 128, 1), 256, 0, stream>>>(
        actA, 0, C, wProj + (size_t)i * C * C, 0, C, hb, 0, C,
        proj_b + (size_t)i * C, M, C, C, 1.0f, 1, 1,
        (i == 0) ? predB : nullptr, (i == 0) ? nullptr : predB);
  }

  // out = h @ fc_out_w^T + fc_out_b   (fp32)
  k_gemm_bf16<true><<<dim3(C / 128, M / 128, 1), 256, 0, stream>>>(
      hb, 0, C, wFcOut, 0, C, out, 0, C, fc_out_b, M, C, C,
      1.0f, 0, 0, nullptr, nullptr);

  // out += pose_enc(S, C)
  const long long n = (long long)M * C;
  k_pose_add<<<(unsigned)((n + 255) / 256), 256, 0, stream>>>(out, S, C, n);
}
